// AttentionLayerO2TwoUpdateNodeGeneral_cross_87548613362080
// MI455X (gfx1250) — compile-verified, matrix operand order checked by reading, not run on hardware
//
#include <hip/hip_runtime.h>

// ---------------- problem constants ----------------
#define NPROT 350
#define NLIG  30
#define NGRPH 32
#define NGN   (NPROT + NLIG)        // 380 nodes per graph
#define NN    (NGRPH * NGN)         // 12160 nodes
#define HID   128
#define HEADS 16
#define DH    8
#define GRBF  20
#define KVW   276

typedef __attribute__((ext_vector_type(16))) __bf16 v16bf;
typedef __attribute__((ext_vector_type(8)))  float  v8f;
typedef __attribute__((ext_vector_type(4)))  unsigned int v4u;

struct bfrag_pair { v4u lo, hi; };   // 32 bytes == one v16bf fragment

// ---------------- scalar helpers ----------------
__device__ inline unsigned short f2bfu(float f) {
    unsigned u = __builtin_bit_cast(unsigned, f);
    unsigned r = u + 0x7FFFu + ((u >> 16) & 1u);   // round-to-nearest-even
    return (unsigned short)(r >> 16);
}
__device__ inline float bfu2f(unsigned short h) {
    unsigned u = ((unsigned)h) << 16;
    return __builtin_bit_cast(float, u);
}
__device__ inline v8f zero8() {
    v8f z = {0.f,0.f,0.f,0.f,0.f,0.f,0.f,0.f};
    return z;
}

// ---------------- WMMA fragment helpers (CDNA5 wave32 layouts) ----------------
// A: 16(M) x 32(K) 16-bit tile from LDS, row-major with leading dim ld (ushorts).
// lane L holds row M=L%16; lanes 0-15: K in {0..7,16..23}, lanes 16-31: {8..15,24..31}.
// Element run e0..7 and e8..15 are each contiguous in memory -> two 16B vector loads.
__device__ inline v16bf load_frag_a(const unsigned short* base, int ld) {
    int lane = threadIdx.x & 31;
    const unsigned short* p = base + (lane & 15) * ld + ((lane >> 4) << 3);
    bfrag_pair pr;
    pr.lo = *(const v4u*)p;           // elems 0..7   (16B aligned)
    pr.hi = *(const v4u*)(p + 16);    // elems 8..15
    return __builtin_bit_cast(v16bf, pr);
}
// B fragments are PRE-PACKED: for each 32x16 K-tile, lane's 16 bf16 elements are
// stored contiguously at base + lane*16. One 32B read per lane, fully coalesced.
__device__ inline v16bf load_frag_b_packed(const unsigned short* base) {
    int lane = threadIdx.x & 31;
    const unsigned short* p = base + lane * 16;
    bfrag_pair pr;
    pr.lo = *(const v4u*)p;
    pr.hi = *(const v4u*)(p + 8);
    return __builtin_bit_cast(v16bf, pr);
}
__device__ inline v8f wmma_bf16(v16bf a, v16bf b, v8f c) {
    return __builtin_amdgcn_wmma_f32_16x16x32_bf16(
        /*neg_a=*/false, a, /*neg_b=*/false, b,
        /*c_mod=*/(short)0, c, /*reuse_a=*/false, /*reuse_b=*/false);
}
// C/D f32 16x16 tile: vgpr v, lane: row = v + 8*(lane>=16), col = lane&15.

// LayerNorm(128) + ReLU over 8 C-layout accumulator tiles -> bf16 LDS [16][128].
__device__ inline void ln_relu_to_lds(v8f* acc, const float* gw, const float* bew,
                                      unsigned short* actW) {
    int lane = threadIdx.x & 31;
    int n  = lane & 15;
    int hf = lane >> 4;
    float s[8], ss[8];
#pragma unroll
    for (int v = 0; v < 8; ++v) { s[v] = 0.f; ss[v] = 0.f; }
#pragma unroll
    for (int ct = 0; ct < 8; ++ct)
#pragma unroll
        for (int v = 0; v < 8; ++v) { float t = acc[ct][v]; s[v] += t; ss[v] += t * t; }
    for (int off = 1; off < 16; off <<= 1)
#pragma unroll
        for (int v = 0; v < 8; ++v) {
            s[v]  += __shfl_xor(s[v],  off, 32);
            ss[v] += __shfl_xor(ss[v], off, 32);
        }
    float mean[8], rstd[8];
#pragma unroll
    for (int v = 0; v < 8; ++v) {
        mean[v] = s[v] * (1.f / 128.f);
        float var = ss[v] * (1.f / 128.f) - mean[v] * mean[v];
        rstd[v] = rsqrtf(var + 1e-5f);
    }
#pragma unroll
    for (int ct = 0; ct < 8; ++ct) {
        int col = ct * 16 + n;
        float g = gw[col], be = bew[col];
#pragma unroll
        for (int v = 0; v < 8; ++v) {
            float y = (acc[ct][v] - mean[v]) * rstd[v] * g + be;
            y = fmaxf(y, 0.f);
            actW[(v + (hf << 3)) * HID + col] = f2bfu(y);
        }
    }
}

// Edge first layer: rbf@W1a (one K=32 WMMA per col-tile) + hoisted dst part +
// (src part + b1), then LN+ReLU -> bf16 activations in actW.
__device__ inline void fused_first_layer(v16bf a_rbf, const unsigned short* rbfP,
                                         const float* cpartSeg, const float* cs,
                                         const float* gw, const float* bew,
                                         int nodeBase, unsigned short* actW) {
    int lane = threadIdx.x & 31;
    int n  = lane & 15;
    int hf = lane >> 4;
    v8f acc[8];
#pragma unroll
    for (int ct = 0; ct < 8; ++ct) {
        v8f c = zero8();
        c = wmma_bf16(a_rbf, load_frag_b_packed(rbfP + ct * 512), c);
        int col = ct * 16 + n;
        float csv = cs[col];
#pragma unroll
        for (int v = 0; v < 8; ++v) {
            int row = v + (hf << 3);
            c[v] += cpartSeg[(size_t)(nodeBase + row) * 768 + col] + csv;
        }
        acc[ct] = c;
    }
    ln_relu_to_lds(acc, gw, bew, actW);
}

// ---------------- workspace layout ----------------
// bf16 (ushort) region at ws start: all weights fragment-packed. Each 32x16
// K-tile block is 512 ushorts; matrix of K x N has (K/32)*(N/16) blocks.
enum : size_t {
    P_HQ1  = 0,                          // 128x128
    P_HQ2  = P_HQ1 + (size_t)HID * HID,
    P_KD   = P_HQ2 + (size_t)HID * HID,  // hk_w1[20:148]
    P_KS   = P_KD  + (size_t)HID * HID,  // hk_w1[148:276]
    P_VD   = P_KS  + (size_t)HID * HID,
    P_VS   = P_VD  + (size_t)HID * HID,
    P_XD   = P_VS  + (size_t)HID * HID,
    P_XS   = P_XD  + (size_t)HID * HID,
    P_W2K  = P_XS  + (size_t)HID * HID,
    P_W2V  = P_W2K + (size_t)HID * HID,
    P_W2X  = P_W2V + (size_t)HID * HID,  // 128x16
    P_NO1  = P_W2X + (size_t)HID * HEADS,// 256x128
    P_NO2  = P_NO1 + (size_t)2 * HID * HID,
    P_RBFK = P_NO2 + (size_t)HID * HID,  // 32x128 zero-padded
    P_RBFV = P_RBFK + (size_t)32 * HID,
    P_RBFX = P_RBFV + (size_t)32 * HID,
    BF16_TOT = P_RBFX + (size_t)32 * HID
};
// float region (elements), after bf16 region
enum : size_t {
    F_Q  = 0,                          // [NN][128] q = MLP_hq(h)
    F_CP = F_Q  + (size_t)NN * HID,    // [NN][768] hoisted first-layer parts
    F_M  = F_CP + (size_t)NN * 768,    // [NN][128] attention output (m)
    F_DX = F_M  + (size_t)NN * HID,    // [NN][4]   coordinate delta
    F_TOT = F_DX + (size_t)NN * 4
};

// ---------------- prep kernels: pack weights into fragment layout ----------------
__global__ void pack_b_kernel(const float* __restrict__ W, unsigned short* __restrict__ dst,
                              int K, int N) {
    int i = blockIdx.x * 256 + threadIdx.x;
    if (i >= K * N) return;
    int e = i & 15;
    int lane = (i >> 4) & 31;
    int blk = i >> 9;              // ks * (N/16) + ct
    int nct = N >> 4;
    int ks = blk / nct, ct = blk - ks * nct;
    int kb = (lane >> 4) << 3;
    int k = (e < 8) ? (kb + e) : (16 + kb + (e - 8));
    int col = ct * 16 + (lane & 15);
    dst[i] = f2bfu(W[(size_t)(32 * ks + k) * N + col]);
}
// rbf weights: rows 0..19 of a [276][128] W1, padded to K=32 with zeros.
__global__ void pack_rbf_kernel(const float* __restrict__ W, unsigned short* __restrict__ dst) {
    int i = blockIdx.x * 256 + threadIdx.x;    // 32*128 packed elems
    if (i >= 32 * HID) return;
    int e = i & 15;
    int lane = (i >> 4) & 31;
    int ct = i >> 9;                            // single kstep
    int kb = (lane >> 4) << 3;
    int k = (e < 8) ? (kb + e) : (16 + kb + (e - 8));
    int col = ct * 16 + (lane & 15);
    dst[i] = (k < GRBF) ? f2bfu(W[(size_t)k * HID + col]) : (unsigned short)0;
}

// ---------------- kernel B: per-node precompute ----------------
// grid 95 x 256 threads; each wave owns a 16-node tile (760 tiles total).
__global__ __launch_bounds__(256) void node_kernel(
    const float* __restrict__ h,
    const float* __restrict__ hq_b1, const float* __restrict__ hq_g,
    const float* __restrict__ hq_be, const float* __restrict__ hq_b2,
    const unsigned short* __restrict__ hq1p, const unsigned short* __restrict__ hq2p,
    const unsigned short* __restrict__ kdp, const unsigned short* __restrict__ ksp,
    const unsigned short* __restrict__ vdp, const unsigned short* __restrict__ vsp,
    const unsigned short* __restrict__ xdp, const unsigned short* __restrict__ xsp,
    float* __restrict__ qbuf, float* __restrict__ cpart,
    float* __restrict__ mbuf, float* __restrict__ dxbuf)
{
    __shared__ unsigned short sm[8][2][16 * HID];   // 64 KB: per-wave h-stage + act
    const int lane = threadIdx.x & 31, wave = threadIdx.x >> 5;
    const int n = lane & 15, hf = lane >> 4;
    const int tile = blockIdx.x * 8 + wave;
    const size_t node0 = (size_t)tile * 16;
    unsigned short* hst = &sm[wave][0][0];
    unsigned short* act = &sm[wave][1][0];

    // stage h tile (16x128) as bf16: float4 loads + packed 8B LDS stores
    for (int idx = lane * 4; idx < 16 * HID; idx += 32 * 4) {
        float4 f = *(const float4*)(h + node0 * HID + idx);
        uint2 pk;
        pk.x = (unsigned)f2bfu(f.x) | ((unsigned)f2bfu(f.y) << 16);
        pk.y = (unsigned)f2bfu(f.z) | ((unsigned)f2bfu(f.w) << 16);
        *(uint2*)(hst + idx) = pk;
    }

    // ---- q = MLP_hq(h) ----
    v8f acc[8];
#pragma unroll
    for (int ct = 0; ct < 8; ++ct) {
        v8f c = zero8();
#pragma unroll
        for (int ks = 0; ks < 4; ++ks)
            c = wmma_bf16(load_frag_a(hst + 32 * ks, HID),
                          load_frag_b_packed(hq1p + (ks * 8 + ct) * 512), c);
        float b = hq_b1[ct * 16 + n];
#pragma unroll
        for (int v = 0; v < 8; ++v) c[v] += b;
        acc[ct] = c;
    }
    ln_relu_to_lds(acc, hq_g, hq_be, act);
#pragma unroll
    for (int ct = 0; ct < 8; ++ct) {
        v8f c = zero8();
#pragma unroll
        for (int ks = 0; ks < 4; ++ks)
            c = wmma_bf16(load_frag_a(act + 32 * ks, HID),
                          load_frag_b_packed(hq2p + (ks * 8 + ct) * 512), c);
        int col = ct * 16 + n;
        float b2 = hq_b2[col];
#pragma unroll
        for (int v = 0; v < 8; ++v)
            qbuf[(node0 + v + (hf << 3)) * HID + col] = c[v] + b2;
    }

    // ---- hoisted first-layer parts: h @ {W1[20:148], W1[148:276]} for hk/hv/xv ----
    const unsigned short* segP[6] = { kdp, ksp, vdp, vsp, xdp, xsp };
#pragma unroll
    for (int seg = 0; seg < 6; ++seg)
#pragma unroll
        for (int ct = 0; ct < 8; ++ct) {
            v8f c = zero8();
#pragma unroll
            for (int ks = 0; ks < 4; ++ks)
                c = wmma_bf16(load_frag_a(hst + 32 * ks, HID),
                              load_frag_b_packed(segP[seg] + (ks * 8 + ct) * 512), c);
            int col = ct * 16 + n;
#pragma unroll
            for (int v = 0; v < 8; ++v)
                cpart[(node0 + v + (hf << 3)) * 768 + seg * HID + col] = c[v];
        }

    // zero accumulation buffers for this tile (vectorized)
    float4 z4{0.f, 0.f, 0.f, 0.f};
    for (int idx = lane * 4; idx < 16 * HID; idx += 32 * 4)
        *(float4*)(mbuf + node0 * HID + idx) = z4;
    for (int idx = lane; idx < 16 * 4; idx += 32) dxbuf[node0 * 4 + idx] = 0.f;
}

// ---------------- kernel C: per-ligand fused edge attention ----------------
// grid = 960 (one WG per ligand node), 256 threads, ~172 KB dynamic LDS.
#define ATTN_LDS_BYTES (352*16*4 /*logits*/ + 352*16*4 /*xvs*/ + 352*4*4 /*relx*/ \
                        + 128*4 /*qs*/ + 3*128*4 /*cs*/ + 64*4 /*dxh*/ \
                        + 352*128*2 /*vst*/ + 8*16*128*2 /*act*/)

__global__ __launch_bounds__(256) void attn_kernel(
    const float* __restrict__ x, const float* __restrict__ qbuf,
    const float* __restrict__ cpart, float* __restrict__ mbuf, float* __restrict__ dxbuf,
    const unsigned short* __restrict__ rbfK, const unsigned short* __restrict__ rbfV,
    const unsigned short* __restrict__ rbfX,
    const unsigned short* __restrict__ w2K, const unsigned short* __restrict__ w2V,
    const unsigned short* __restrict__ w2X,
    const float* __restrict__ kb1, const float* __restrict__ kg,
    const float* __restrict__ kbe, const float* __restrict__ kb2,
    const float* __restrict__ vb1, const float* __restrict__ vg,
    const float* __restrict__ vbe, const float* __restrict__ vb2,
    const float* __restrict__ xb1, const float* __restrict__ xg,
    const float* __restrict__ xbe, const float* __restrict__ xb2)
{
    extern __shared__ char smem[];
    float* logits = (float*)smem;             // [352][16]
    float* xvs    = logits + 352 * 16;        // [352][16]
    float* relx   = xvs + 352 * 16;           // [352][4]
    float* qs     = relx + 352 * 4;           // [128]
    float* csK    = qs + 128;                 // [128] src-part + b1
    float* csV    = csK + 128;
    float* csX    = csV + 128;
    float* dxh    = csX + 128;                // [16][4]
    unsigned short* vst    = (unsigned short*)(dxh + 64);   // [352][128] bf16
    unsigned short* actAll = vst + 352 * 128;               // 8 x [16][128]

    const int tid = threadIdx.x, lane = tid & 31, wave = tid >> 5;
    const int n = lane & 15, hf = lane >> 4;
    const int bg = blockIdx.x / NLIG, li = blockIdx.x % NLIG;
    const int lig = bg * NGN + NPROT + li;
    const int pbase = bg * NGN;
    unsigned short* actW = actAll + wave * (16 * HID);

    for (int c = tid; c < HID; c += 256) {
        qs[c]  = qbuf[(size_t)lig * HID + c];
        csK[c] = cpart[(size_t)lig * 768 + 128 + c] + kb1[c];
        csV[c] = cpart[(size_t)lig * 768 + 384 + c] + vb1[c];
        csX[c] = cpart[(size_t)lig * 768 + 640 + c] + xb1[c];
    }
    const float xl0 = x[lig * 3 + 0], xl1 = x[lig * 3 + 1], xl2 = x[lig * 3 + 2];
    __syncthreads();

    const float dlt = 10.f / 19.f;
    const float coeff = -0.5f / (dlt * dlt);

    // ---- phase 1: per 16-protein tile, fused MLPs via WMMA ----
    for (int pt = wave; pt < 22; pt += 8) {
        const int p0 = pt * 16;
        // prefetch next tile's hoisted parts (CDNA5 global_prefetch path)
        if (pt + 8 < 22) {
            const float* nxt = cpart + (size_t)(pbase + (pt + 8) * 16 + (lane >> 1)) * 768
                                     + (lane & 1) * 384;
            __builtin_prefetch(nxt, 0, 1);
        }
        {   // rbf features -> actW[16][32] (zero-padded K 20..31)
            int r = n;
            int pI = p0 + r;
            int pn = pbase + (pI < NPROT ? pI : NPROT - 1);
            float rx = xl0 - x[pn * 3 + 0];
            float ry = xl1 - x[pn * 3 + 1];
            float rz = xl2 - x[pn * 3 + 2];
            float dist = sqrtf(rx * rx + ry * ry + rz * rz);
            if (hf == 0) { relx[pI * 4 + 0] = rx; relx[pI * 4 + 1] = ry; relx[pI * 4 + 2] = rz; }
#pragma unroll
            for (int j = 0; j < 8; ++j) {
                int f0 = hf * 16 + 2 * j;
                float v0 = 0.f, v1 = 0.f;
                if (f0 < GRBF)     { float d = dist - dlt * (float)f0;       v0 = expf(coeff * d * d); }
                if (f0 + 1 < GRBF) { float d = dist - dlt * (float)(f0 + 1); v1 = expf(coeff * d * d); }
                unsigned pk = (unsigned)f2bfu(v0) | ((unsigned)f2bfu(v1) << 16);
                *(unsigned*)(actW + r * 32 + f0) = pk;
            }
        }
        v16bf a_rbf = load_frag_a(actW, 32);

        // ---- K branch: logits ----
        fused_first_layer(a_rbf, rbfK, cpart + 0, csK, kg, kbe, pbase + p0, actW);
#pragma unroll
        for (int ct = 0; ct < 8; ++ct) {
            v8f c = zero8();
#pragma unroll
            for (int ks = 0; ks < 4; ++ks)
                c = wmma_bf16(load_frag_a(actW + 32 * ks, HID),
                              load_frag_b_packed(w2K + (ks * 8 + ct) * 512), c);
            int col = ct * 16 + n;
            float b2v = kb2[col];
            float part[8];
#pragma unroll
            for (int v = 0; v < 8; ++v) part[v] = (c[v] + b2v) * qs[col];
            for (int off = 1; off < 8; off <<= 1)
#pragma unroll
                for (int v = 0; v < 8; ++v) part[v] += __shfl_xor(part[v], off, 32);
            if ((n & 7) == 0) {
                int head = col >> 3;                      // = ct*2 + n/8
#pragma unroll
                for (int v = 0; v < 8; ++v) {
                    int p = p0 + v + (hf << 3);
                    logits[p * 16 + head] = part[v] * 0.3535533906f;  // 1/sqrt(8)
                }
            }
        }

        // ---- V branch ----
        fused_first_layer(a_rbf, rbfV, cpart + 256, csV, vg, vbe, pbase + p0, actW);
#pragma unroll
        for (int ct = 0; ct < 8; ++ct) {
            v8f c = zero8();
#pragma unroll
            for (int ks = 0; ks < 4; ++ks)
                c = wmma_bf16(load_frag_a(actW + 32 * ks, HID),
                              load_frag_b_packed(w2V + (ks * 8 + ct) * 512), c);
            int col = ct * 16 + n;
            float b2v = vb2[col];
#pragma unroll
            for (int v = 0; v < 8; ++v)
                vst[(size_t)(p0 + v + (hf << 3)) * HID + col] = f2bfu(c[v] + b2v);
        }

        // ---- XV branch (128 -> 16) ----
        fused_first_layer(a_rbf, rbfX, cpart + 512, csX, xg, xbe, pbase + p0, actW);
        {
            v8f c = zero8();
#pragma unroll
            for (int ks = 0; ks < 4; ++ks)
                c = wmma_bf16(load_frag_a(actW + 32 * ks, HID),
                              load_frag_b_packed(w2X + ks * 512), c);
            float b2v = xb2[n];
#pragma unroll
            for (int v = 0; v < 8; ++v)
                xvs[(p0 + v + (hf << 3)) * 16 + n] = c[v] + b2v;
        }
    }
    __syncthreads();

    // ---- phase 2: softmax over 350 neighbors + aggregation; 2 heads per wave ----
    const int head = wave * 2 + hf;
    float mx = -1e30f;
    for (int p = n; p < NPROT; p += 16) mx = fmaxf(mx, logits[p * 16 + head]);
    for (int off = 1; off < 16; off <<= 1) mx = fmaxf(mx, __shfl_xor(mx, off, 32));
    float ssum = 0.f;
    for (int p = n; p < NPROT; p += 16) ssum += expf(logits[p * 16 + head] - mx);
    for (int off = 1; off < 16; off <<= 1) ssum += __shfl_xor(ssum, off, 32);
    const float inv = 1.f / ssum;

    float macc[8] = {0.f,0.f,0.f,0.f,0.f,0.f,0.f,0.f};
    float dxa0 = 0.f, dxa1 = 0.f, dxa2 = 0.f;
    for (int p = n; p < NPROT; p += 16) {
        float al = expf(logits[p * 16 + head] - mx) * inv;
        v4u vv = *(const v4u*)(vst + (size_t)p * HID + head * DH);   // one ds_load_b128
#pragma unroll
        for (int d = 0; d < 8; ++d) {
            unsigned short us = (unsigned short)((vv[d >> 1] >> ((d & 1) * 16)) & 0xFFFFu);
            macc[d] += al * bfu2f(us);
        }
        float axv = al * xvs[p * 16 + head];
        dxa0 += axv * relx[p * 4 + 0];
        dxa1 += axv * relx[p * 4 + 1];
        dxa2 += axv * relx[p * 4 + 2];
    }
    for (int off = 1; off < 16; off <<= 1) {
#pragma unroll
        for (int d = 0; d < 8; ++d) macc[d] += __shfl_xor(macc[d], off, 32);
        dxa0 += __shfl_xor(dxa0, off, 32);
        dxa1 += __shfl_xor(dxa1, off, 32);
        dxa2 += __shfl_xor(dxa2, off, 32);
    }
    if (n == 0) {
#pragma unroll
        for (int d = 0; d < 8; ++d) mbuf[(size_t)lig * HID + head * DH + d] = macc[d];
        dxh[head * 4 + 0] = dxa0; dxh[head * 4 + 1] = dxa1; dxh[head * 4 + 2] = dxa2;
    }
    __syncthreads();
    if (tid < 3) {
        float s = 0.f;
        for (int hh = 0; hh < HEADS; ++hh) s += dxh[hh * 4 + tid];
        dxbuf[(size_t)lig * 4 + tid] = s * (1.f / 16.f);
    }
}

// ---------------- kernel D: output MLP + residual + coordinate update ----------------
#define OUT_LDS_BYTES (8*16*256*2 /*stage*/ + 8*16*128*2 /*act*/)
__global__ __launch_bounds__(256) void out_kernel(
    const float* __restrict__ h, const float* __restrict__ x, const int* __restrict__ mask,
    const float* __restrict__ mbuf, const float* __restrict__ dxbuf,
    const unsigned short* __restrict__ no1p, const unsigned short* __restrict__ no2p,
    const float* __restrict__ b1, const float* __restrict__ g,
    const float* __restrict__ be, const float* __restrict__ b2,
    float* __restrict__ hout, float* __restrict__ xout)
{
    extern __shared__ char smemD[];
    unsigned short* stageAll = (unsigned short*)smemD;      // 8 x [16][256]
    unsigned short* actAll   = stageAll + 8 * 16 * 256;     // 8 x [16][128]
    const int lane = threadIdx.x & 31, wave = threadIdx.x >> 5;
    const int n = lane & 15, hf = lane >> 4;
    const int tile = blockIdx.x * 8 + wave;
    const size_t node0 = (size_t)tile * 16;
    unsigned short* st  = stageAll + wave * (16 * 256);
    unsigned short* act = actAll + wave * (16 * HID);

    // stage concat(m, h[mask_row]) as bf16 [16][256], vectorized
    for (int idx = lane * 4; idx < 16 * 256; idx += 32 * 4) {
        int r = idx >> 8, c = idx & 255;
        float4 f;
        if (c < 128) f = *(const float4*)(mbuf + (node0 + r) * HID + c);
        else {
            int sel = mask[node0 + r];                    // faithful: gathers h row 0/1
            f = *(const float4*)(h + (size_t)sel * HID + (c - 128));
        }
        uint2 pk;
        pk.x = (unsigned)f2bfu(f.x) | ((unsigned)f2bfu(f.y) << 16);
        pk.y = (unsigned)f2bfu(f.z) | ((unsigned)f2bfu(f.w) << 16);
        *(uint2*)(st + idx) = pk;
    }

    v8f acc[8];
#pragma unroll
    for (int ct = 0; ct < 8; ++ct) {
        v8f c = zero8();
#pragma unroll
        for (int ks = 0; ks < 8; ++ks)
            c = wmma_bf16(load_frag_a(st + 32 * ks, 256),
                          load_frag_b_packed(no1p + (ks * 8 + ct) * 512), c);
        float b = b1[ct * 16 + n];
#pragma unroll
        for (int v = 0; v < 8; ++v) c[v] += b;
        acc[ct] = c;
    }
    ln_relu_to_lds(acc, g, be, act);
#pragma unroll
    for (int ct = 0; ct < 8; ++ct) {
        v8f c = zero8();
#pragma unroll
        for (int ks = 0; ks < 4; ++ks)
            c = wmma_bf16(load_frag_a(act + 32 * ks, HID),
                          load_frag_b_packed(no2p + (ks * 8 + ct) * 512), c);
        int col = ct * 16 + n;
        float b2v = b2[col];
#pragma unroll
        for (int v = 0; v < 8; ++v) {
            size_t nd = node0 + v + (hf << 3);
            hout[nd * HID + col] = c[v] + b2v + h[nd * HID + col];   // residual
        }
    }
    for (int idx = lane; idx < 48; idx += 32) {
        int r = idx / 3, k = idx % 3;
        size_t nd = node0 + r;
        xout[nd * 3 + k] = x[nd * 3 + k] + dxbuf[nd * 4 + k];
    }
}

// ---------------- host entry ----------------
extern "C" void kernel_launch(void* const* d_in, const int* in_sizes, int n_in,
                              void* d_out, int out_size, void* d_ws, size_t ws_size,
                              hipStream_t stream) {
    (void)in_sizes; (void)n_in; (void)out_size; (void)ws_size;
    const float* h    = (const float*)d_in[0];
    const float* x    = (const float*)d_in[1];
    const int*   mask = (const int*)d_in[4];
    const float* hk_w1 = (const float*)d_in[5];
    const float* hk_b1 = (const float*)d_in[6];
    const float* hk_g  = (const float*)d_in[7];
    const float* hk_be = (const float*)d_in[8];
    const float* hk_w2 = (const float*)d_in[9];
    const float* hk_b2 = (const float*)d_in[10];
    const float* hv_w1 = (const float*)d_in[11];
    const float* hv_b1 = (const float*)d_in[12];
    const float* hv_g  = (const float*)d_in[13];
    const float* hv_be = (const float*)d_in[14];
    const float* hv_w2 = (const float*)d_in[15];
    const float* hv_b2 = (const float*)d_in[16];
    const float* xv_w1 = (const float*)d_in[17];
    const float* xv_b1 = (const float*)d_in[18];
    const float* xv_g  = (const float*)d_in[19];
    const float* xv_be = (const float*)d_in[20];
    const float* xv_w2 = (const float*)d_in[21];
    const float* xv_b2 = (const float*)d_in[22];
    const float* hq_w1 = (const float*)d_in[23];
    const float* hq_b1 = (const float*)d_in[24];
    const float* hq_g  = (const float*)d_in[25];
    const float* hq_be = (const float*)d_in[26];
    const float* hq_w2 = (const float*)d_in[27];
    const float* hq_b2 = (const float*)d_in[28];
    const float* no_w1 = (const float*)d_in[29];
    const float* no_b1 = (const float*)d_in[30];
    const float* no_g  = (const float*)d_in[31];
    const float* no_be = (const float*)d_in[32];
    const float* no_w2 = (const float*)d_in[33];
    const float* no_b2 = (const float*)d_in[34];

    unsigned short* bfb = (unsigned short*)d_ws;
    unsigned short* hq1p = bfb + P_HQ1;
    unsigned short* hq2p = bfb + P_HQ2;
    unsigned short* kdp  = bfb + P_KD;
    unsigned short* ksp  = bfb + P_KS;
    unsigned short* vdp  = bfb + P_VD;
    unsigned short* vsp  = bfb + P_VS;
    unsigned short* xdp  = bfb + P_XD;
    unsigned short* xsp  = bfb + P_XS;
    unsigned short* w2Kp = bfb + P_W2K;
    unsigned short* w2Vp = bfb + P_W2V;
    unsigned short* w2Xp = bfb + P_W2X;
    unsigned short* no1p = bfb + P_NO1;
    unsigned short* no2p = bfb + P_NO2;
    unsigned short* rbfK = bfb + P_RBFK;
    unsigned short* rbfV = bfb + P_RBFV;
    unsigned short* rbfX = bfb + P_RBFX;
    float* fbase = (float*)(bfb + BF16_TOT);
    float* qbuf  = fbase + F_Q;
    float* cpart = fbase + F_CP;
    float* mbuf  = fbase + F_M;
    float* dxbuf = fbase + F_DX;

    auto pk = [&](const float* s, unsigned short* d, int K, int N) {
        pack_b_kernel<<<(K * N + 255) / 256, 256, 0, stream>>>(s, d, K, N);
    };
    pk(hq_w1, hq1p, HID, HID);
    pk(hq_w2, hq2p, HID, HID);
    pk(hk_w1 + (size_t)GRBF * HID,        kdp, HID, HID);
    pk(hk_w1 + (size_t)(GRBF + HID) * HID, ksp, HID, HID);
    pk(hv_w1 + (size_t)GRBF * HID,        vdp, HID, HID);
    pk(hv_w1 + (size_t)(GRBF + HID) * HID, vsp, HID, HID);
    pk(xv_w1 + (size_t)GRBF * HID,        xdp, HID, HID);
    pk(xv_w1 + (size_t)(GRBF + HID) * HID, xsp, HID, HID);
    pk(hk_w2, w2Kp, HID, HID);
    pk(hv_w2, w2Vp, HID, HID);
    pk(xv_w2, w2Xp, HID, HEADS);
    pk(no_w1, no1p, 2 * HID, HID);
    pk(no_w2, no2p, HID, HID);
    pack_rbf_kernel<<<16, 256, 0, stream>>>(hk_w1, rbfK);
    pack_rbf_kernel<<<16, 256, 0, stream>>>(hv_w1, rbfV);
    pack_rbf_kernel<<<16, 256, 0, stream>>>(xv_w1, rbfX);

    node_kernel<<<NN / (16 * 8), 256, 0, stream>>>(
        h, hq_b1, hq_g, hq_be, hq_b2, hq1p, hq2p,
        kdp, ksp, vdp, vsp, xdp, xsp,
        qbuf, cpart, mbuf, dxbuf);

    attn_kernel<<<NGRPH * NLIG, 256, ATTN_LDS_BYTES, stream>>>(
        x, qbuf, cpart, mbuf, dxbuf,
        rbfK, rbfV, rbfX, w2Kp, w2Vp, w2Xp,
        hk_b1, hk_g, hk_be, hk_b2,
        hv_b1, hv_g, hv_be, hv_b2,
        xv_b1, xv_g, xv_be, xv_b2);

    float* hout = (float*)d_out;
    float* xout = hout + (size_t)NN * HID;
    out_kernel<<<NN / (16 * 8), 256, OUT_LDS_BYTES, stream>>>(
        h, x, mask, mbuf, dxbuf, no1p, no2p, no_b1, no_g, no_be, no_b2, hout, xout);
}
